// AttnSageGCN_68951404970163
// MI455X (gfx1250) — compile-verified, matrix operand order checked by reading, not run on hardware
//
#include <hip/hip_runtime.h>

typedef __attribute__((ext_vector_type(16))) __bf16 v16bf;
typedef __attribute__((ext_vector_type(8)))  float  v8f;

#define D       128
#define H2      256
#define KNB     32
#define HIDDEN  128
#define HEADS   4
#define DPH     32
#define SCALE_F 0.17677669529663687f   // 32^-0.5

#define TPB     256
#define NPB     8      // nodes per block (attn kernel)
#define ROWS1   32     // src rows per block (proj kernel)

#define W_STRIDE  136  // padded bf16 row stride (68 dwords -> conflict-free b128 frag loads)
#define NB_STRIDE 136
#define KV_STRIDE 257  // odd dword stride -> conflict-free column access

// ---- CDNA5 async global->LDS copy (16B), tracked by ASYNCcnt ----
__device__ __forceinline__ void async_ld16(void* lds_ptr, const void* gptr) {
    // VGLOBAL GV mode: VDST = LDS byte address, VADDR = 64-bit global address.
    // Generic LDS pointers keep the LDS offset in the low 32 bits (ISA 10.2).
    asm volatile("global_load_async_to_lds_b128 %0, %1, off"
                 :
                 : "v"((uint32_t)(uintptr_t)lds_ptr),
                   "v"((uint64_t)(uintptr_t)gptr)
                 : "memory");
}
__device__ __forceinline__ void wait_async0() {
    asm volatile("s_wait_asynccnt 0x0" ::: "memory");
}

// Load a 16x32 bf16 WMMA A/B fragment from an LDS row (ISA 7.12.2 layout):
// lane holds one row; elems 0..7 = K[kbase..kbase+7], elems 8..15 = K[kbase+16..kbase+23]
__device__ __forceinline__ v16bf frag_ld(const __bf16* rowbase, int kbase) {
    v16bf f;
#pragma unroll
    for (int i = 0; i < 8; ++i) {
        f[i]     = rowbase[kbase + i];
        f[i + 8] = rowbase[kbase + 16 + i];
    }
    return f;
}

// ---------------- kernel 0: weight convert/transpose to bf16 ----------------
__global__ __launch_bounds__(TPB) void convert_weights(
    const float* __restrict__ wq, const float* __restrict__ wkv,
    const float* __restrict__ wo, const float* __restrict__ w_self,
    __bf16* __restrict__ wkvT, __bf16* __restrict__ wcatT, __bf16* __restrict__ woB)
{
    int i = blockIdx.x * TPB + threadIdx.x;
    if (i < H2 * D) {                         // wkvT[n][k] = wkv[k][n]
        int n = i >> 7, k = i & 127;
        wkvT[i] = (__bf16)wkv[k * H2 + n];
    } else if (i < 2 * H2 * D) {              // wcatT[n][k] = [wq | w_self] transposed
        int j = i - H2 * D;
        int n = j >> 7, k = j & 127;
        float v = (n < HIDDEN) ? wq[k * HIDDEN + n] : w_self[k * HIDDEN + (n - HIDDEN)];
        wcatT[j] = (__bf16)v;
    } else if (i < 2 * H2 * D + HIDDEN * HIDDEN) {
        int j = i - 2 * H2 * D;
        woB[j] = (__bf16)wo[j];               // row-major copy
    }
}

// ---------------- kernel 1: q = src@wq+bq ; self = src@w_self (WMMA) ----------------
__global__ __launch_bounds__(TPB) void proj_qself(
    const float* __restrict__ src, const float* __restrict__ bq,
    const __bf16* __restrict__ wcatT,
    float* __restrict__ qbuf, float* __restrict__ selfbuf)
{
    __shared__ __align__(16) __bf16 sW[H2 * W_STRIDE];
    __shared__ __align__(16) __bf16 sS[ROWS1 * NB_STRIDE];
    __shared__ float sB2[H2];   // [bq | zeros]

    const int tid = threadIdx.x;
    const int lane = tid & 31, wv = tid >> 5;
    const int lr = lane & 15, khalf = lane >> 4;

    // weights -> LDS via async DMA (16B chunks, padded rows)
    for (int c = tid; c < H2 * (D / 8); c += TPB) {
        int row = c >> 4, col = (c & 15) << 3;
        async_ld16(&sW[row * W_STRIDE + col], &wcatT[row * D + col]);
    }
    // src f32 -> bf16 LDS (needs conversion: VALU path)
    const float4* s4 = (const float4*)(src + (size_t)blockIdx.x * ROWS1 * D);
    for (int c = tid; c < ROWS1 * (D / 4); c += TPB) {
        float4 v = s4[c];
        int row = c >> 5, col = (c & 31) << 2;
        __bf16* d = &sS[row * NB_STRIDE + col];
        d[0] = (__bf16)v.x; d[1] = (__bf16)v.y; d[2] = (__bf16)v.z; d[3] = (__bf16)v.w;
    }
    if (tid < H2) sB2[tid] = (tid < HIDDEN) ? bq[tid] : 0.0f;
    wait_async0();
    __syncthreads();

    const int m0 = (wv & 1) * 16;
    const int ngrp = wv >> 1;
    const int isQ = __builtin_amdgcn_readfirstlane((ngrp < 2) ? 1 : 0);  // wave-uniform
    float* const dbase = isQ ? qbuf : selfbuf;

    v16bf a[4];
#pragma unroll
    for (int kc = 0; kc < 4; ++kc)
        a[kc] = frag_ld(&sS[(m0 + lr) * NB_STRIDE], kc * 32 + khalf * 8);

#pragma unroll
    for (int j = 0; j < 4; ++j) {
        const int n0 = (ngrp * 4 + j) * 16;
        v8f acc = {};
#pragma unroll
        for (int kc = 0; kc < 4; ++kc) {
            v16bf b = frag_ld(&sW[(n0 + lr) * W_STRIDE], kc * 32 + khalf * 8);
            acc = __builtin_amdgcn_wmma_f32_16x16x32_bf16(
                false, a[kc], false, b, (short)0, acc, false, false);
        }
        const int n = n0 + lr;
        const float bias = sB2[n];
        const int ncol = n & 127;
        const int grow0 = blockIdx.x * ROWS1 + m0 + khalf * 8;
#pragma unroll
        for (int r = 0; r < 8; ++r)
            dbase[(size_t)(grow0 + r) * HIDDEN + ncol] = acc[r] + bias;
    }
}

// ---------------- kernel 2: fused KV-proj + attention + out-proj + relu ----------------
__global__ __launch_bounds__(TPB) void attn_fused(
    const float* __restrict__ neighbors, const float* __restrict__ bkv,
    const float* __restrict__ bo,
    const __bf16* __restrict__ wkvT, const __bf16* __restrict__ woB,
    const float* __restrict__ qbuf, const float* __restrict__ selfbuf,
    float* __restrict__ out)
{
    __shared__ __align__(16) __bf16 sW[H2 * W_STRIDE];       // wkvT    68 KB
    __shared__ __align__(16) __bf16 sN[KNB * NB_STRIDE];     // nbrs     8.5 KB
    __shared__ __align__(16) __bf16 sWo[HIDDEN * HIDDEN];    // wo      32 KB
    __shared__ float sKV[KNB * KV_STRIDE];                   //         32.9 KB
    __shared__ float sBkv[H2];
    __shared__ float sBo[HIDDEN];
    __shared__ float sQ[HIDDEN];
    __shared__ float sP[HEADS * KNB];
    __shared__ float sOut[HIDDEN];

    const int tid = threadIdx.x;
    const int lane = tid & 31, wv = tid >> 5;
    const int lr = lane & 15, khalf = lane >> 4;

    // weight tiles -> LDS via async DMA (no conversion needed)
    for (int c = tid; c < H2 * (D / 8); c += TPB) {
        int row = c >> 4, col = (c & 15) << 3;
        async_ld16(&sW[row * W_STRIDE + col], &wkvT[row * D + col]);
    }
    for (int c = tid; c < HIDDEN * HIDDEN / 8; c += TPB)
        async_ld16(&((uint4*)sWo)[c], &((const uint4*)woB)[c]);
    if (tid < H2)     sBkv[tid] = bkv[tid];
    if (tid < HIDDEN) sBo[tid]  = bo[tid];
    wait_async0();
    __syncthreads();

    for (int ni = 0; ni < NPB; ++ni) {
        const int node = blockIdx.x * NPB + ni;

        // ---- stage neighbors (f32 -> bf16) + q; prefetch next node's tile ----
        const float4* n4 = (const float4*)(neighbors + (size_t)node * KNB * D);
        for (int c = tid; c < KNB * (D / 4); c += TPB) {
            float4 v = n4[c];
            int row = c >> 5, col = (c & 31) << 2;
            __bf16* d = &sN[row * NB_STRIDE + col];
            d[0] = (__bf16)v.x; d[1] = (__bf16)v.y; d[2] = (__bf16)v.z; d[3] = (__bf16)v.w;
        }
        if (ni + 1 < NPB)   // 256 threads x 64B = next 16 KB tile
            __builtin_prefetch(neighbors + (size_t)(node + 1) * KNB * D + tid * 16, 0, 0);
        if (tid < HIDDEN) sQ[tid] = qbuf[node * HIDDEN + tid];
        __syncthreads();

        // ---- KV = Nbr(32x128) @ Wkv(128x256) via bf16 WMMA ----
        const int m0 = (wv & 1) * 16;
        const int ngrp = wv >> 1;
        v16bf a[4];
#pragma unroll
        for (int kc = 0; kc < 4; ++kc)
            a[kc] = frag_ld(&sN[(m0 + lr) * NB_STRIDE], kc * 32 + khalf * 8);
#pragma unroll
        for (int j = 0; j < 4; ++j) {
            const int n0 = (ngrp * 4 + j) * 16;
            v8f acc = {};
#pragma unroll
            for (int kc = 0; kc < 4; ++kc) {
                v16bf b = frag_ld(&sW[(n0 + lr) * W_STRIDE], kc * 32 + khalf * 8);
                acc = __builtin_amdgcn_wmma_f32_16x16x32_bf16(
                    false, a[kc], false, b, (short)0, acc, false, false);
            }
            const int n = n0 + lr;
            const int mb = m0 + khalf * 8;
#pragma unroll
            for (int r = 0; r < 8; ++r)
                sKV[(mb + r) * KV_STRIDE + n] = acc[r] + sBkv[n];
        }
        __syncthreads();

        // ---- scores + softmax: wave = head, lane = neighbor ----
        if (wv < HEADS) {
            const int h = wv, k = lane;
            const float* kr = &sKV[k * KV_STRIDE + h * DPH];
            const float* qh = &sQ[h * DPH];
            float s = 0.f;
#pragma unroll
            for (int dd = 0; dd < DPH; ++dd) s += qh[dd] * kr[dd];
            s *= SCALE_F;
            float mx = s;
#pragma unroll
            for (int off = 16; off > 0; off >>= 1) mx = fmaxf(mx, __shfl_xor(mx, off, 32));
            float p = __expf(s - mx);
            float sum = p;
#pragma unroll
            for (int off = 16; off > 0; off >>= 1) sum += __shfl_xor(sum, off, 32);
            sP[h * KNB + k] = p / sum;
        }
        __syncthreads();

        // ---- out = P @ V: wave = head, lane = dim ----
        if (wv < HEADS) {
            const int h = wv, dd = lane;
            float o = 0.f;
#pragma unroll
            for (int k = 0; k < KNB; ++k)
                o += sP[h * KNB + k] * sKV[k * KV_STRIDE + HIDDEN + h * DPH + dd];
            sOut[h * DPH + dd] = o;
        }
        __syncthreads();

        // ---- final = relu(self + out@wo + bo) ----
        if (tid < HIDDEN) {
            const int n = tid;
            float acc = selfbuf[node * HIDDEN + n] + sBo[n];
#pragma unroll 8
            for (int i = 0; i < HIDDEN; ++i)
                acc += sOut[i] * (float)sWo[i * HIDDEN + n];
            out[node * HIDDEN + n] = fmaxf(acc, 0.f);
        }
        __syncthreads();
    }
}

extern "C" void kernel_launch(void* const* d_in, const int* in_sizes, int n_in,
                              void* d_out, int out_size, void* d_ws, size_t ws_size,
                              hipStream_t stream) {
    (void)n_in; (void)out_size; (void)ws_size;
    const float* src       = (const float*)d_in[0];
    const float* neighbors = (const float*)d_in[1];
    const float* wq        = (const float*)d_in[2];
    const float* bq        = (const float*)d_in[3];
    const float* wkv       = (const float*)d_in[4];
    const float* bkv       = (const float*)d_in[5];
    const float* wo        = (const float*)d_in[6];
    const float* bo        = (const float*)d_in[7];
    const float* w_self    = (const float*)d_in[8];
    float* out = (float*)d_out;
    const int B = in_sizes[0] / D;

    char* ws = (char*)d_ws;
    __bf16* wkvT  = (__bf16*)(ws);                 // 64 KB
    __bf16* wcatT = (__bf16*)(ws + 65536);         // 64 KB
    __bf16* woB   = (__bf16*)(ws + 131072);        // 32 KB
    float*  qbuf  = (float*)(ws + 163840);                                  // B*128 f32
    float*  selfb = (float*)(ws + 163840 + (size_t)B * HIDDEN * sizeof(float)); // B*128 f32

    const int totalConv = 2 * H2 * D + HIDDEN * HIDDEN;
    convert_weights<<<(totalConv + TPB - 1) / TPB, TPB, 0, stream>>>(
        wq, wkv, wo, w_self, wkvT, wcatT, woB);
    proj_qself<<<B / ROWS1, TPB, 0, stream>>>(src, bq, wcatT, qbuf, selfb);
    attn_fused<<<B / NPB, TPB, 0, stream>>>(
        neighbors, bkv, bo, wkvT, woB, qbuf, selfb, out);
}
